// GraphAutoEncoder_11879879541076
// MI455X (gfx1250) — matrix operand order repaired and checked by reading, not compiled, not run on hardware
//
#include <hip/hip_runtime.h>

// ---------------------------------------------------------------------------
// GATv2 graph autoencoder forward for gfx1250 (MI455X, wave32, WMMA).
// GEMMs (N x 128 @ 128 x 128) run on v_wmma_f32_16x16x32_bf16 (f32 accum).
// Segment softmax / scatter use native f32 global atomics (L2-resident).
// ---------------------------------------------------------------------------

#define BATCH 8
#define NN    8192
#define NE    49152
#define ET    (NE + NN)          // 57344 edges incl. self loops (=8*7168)
#define HD    128
#define BIG   ((size_t)BATCH * NN * HD)
#define KP    136                // padded LDS row stride (bf16 elems)

typedef __attribute__((ext_vector_type(16))) __bf16 v16bf;
typedef __attribute__((ext_vector_type(8)))  float  v8f;

__device__ __forceinline__ void atomicMaxF(float* addr, float v) {
  // IEEE-ordered max via integer atomics (init must be -inf).
  if (v >= 0.0f) atomicMax((int*)addr, __float_as_int(v));
  else           atomicMin((unsigned int*)addr, __float_as_uint(v));
}

// -------------------------------- utility ----------------------------------
__global__ __launch_bounds__(256) void fill_kernel(float* __restrict__ p, float v, size_t n) {
  size_t i = (size_t)blockIdx.x * 256 + threadIdx.x;
  if (i < n) p[i] = v;
}

// ------------------------- encoder stage 1 (K=5) ---------------------------
__global__ __launch_bounds__(256) void h1_kernel(const float* __restrict__ obs,
                                                 const float* __restrict__ ew1,
                                                 const float* __restrict__ eb1,
                                                 float* __restrict__ H1) {
  size_t i = (size_t)blockIdx.x * 256 + threadIdx.x;      // over B*N*128
  if (i >= BIG) return;
  int h = (int)(i & 127);
  size_t bn = i >> 7;
  const float* o = obs + bn * 5;
  float s = eb1[h];
#pragma unroll
  for (int k = 0; k < 5; ++k) s += o[k] * ew1[h * 5 + k];
  H1[i] = s > 0.f ? s : 0.f;
}

// ------------------- WMMA GEMM: OUT = act(X @ W.T + b) ---------------------
// X: (N,128) f32 per graph, W: (128,128) row-major, OUT: (N,128) f32.
// Block = 256 threads = 8 waves; block tile = 128 rows x 128 cols.
// Each wave: 16 rows x 128 cols = 8 WMMA tiles, K=128 in 4 steps of 32.
__global__ __launch_bounds__(256) void gemm128_kernel(const float* __restrict__ X,
                                                      const float* __restrict__ W,
                                                      const float* __restrict__ bias,
                                                      float* __restrict__ OUT,
                                                      int relu_flag) {
  __shared__ __bf16 As[128 * KP];   // activations, row-major [row][k]
  __shared__ __bf16 Bs[128 * KP];   // weights,     row-major [col][k] (== W)
  const int b    = blockIdx.y;
  const int row0 = blockIdx.x * 128;
  const float* Xb = X + (size_t)b * NN * HD;
  float*       Ob = OUT + (size_t)b * NN * HD;
  const int t = threadIdx.x;
  for (int i = t; i < 128 * 128; i += 256) {
    int r = i >> 7, k = i & 127;
    As[r * KP + k] = (__bf16)Xb[(size_t)(row0 + r) * HD + k];
    Bs[r * KP + k] = (__bf16)W[i];
  }
  __syncthreads();

  const int wave = t >> 5;
  const int lane = t & 31;
  const int rl   = (wave << 4) + (lane & 15);   // A-frag row in block tile
  const int ka   = (lane < 16) ? 0 : 8;         // A k-base  (ISA 16b A layout)
  const int kb   = (lane < 16) ? 0 : 8;         // B k-base  (ISA 16b B layout)

  v8f acc[8];
#pragma unroll
  for (int nt = 0; nt < 8; ++nt) acc[nt] = (v8f){0.f,0.f,0.f,0.f,0.f,0.f,0.f,0.f};

#pragma unroll
  for (int kt = 0; kt < 4; ++kt) {
    v16bf a;
    const __bf16* pa = &As[rl * KP + kt * 32 + ka];
#pragma unroll
    for (int j = 0; j < 8; ++j) { a[j] = pa[j]; a[8 + j] = pa[16 + j]; }
#pragma unroll
    for (int nt = 0; nt < 8; ++nt) {
      v16bf bb;
      const __bf16* pb = &Bs[(nt * 16 + (lane & 15)) * KP + kt * 32 + kb];
#pragma unroll
      for (int j = 0; j < 8; ++j) { bb[j] = pb[j]; bb[8 + j] = pb[16 + j]; }
      acc[nt] = __builtin_amdgcn_wmma_f32_16x16x32_bf16(
          false, a, false, bb, (short)0, acc[nt], false, false);
    }
  }

  const int rowBase = row0 + (wave << 4) + ((lane >= 16) ? 8 : 0);
#pragma unroll
  for (int nt = 0; nt < 8; ++nt) {
    int col = nt * 16 + (lane & 15);
    float bv = bias[col];
#pragma unroll
    for (int r = 0; r < 8; ++r) {
      float v = acc[nt][r] + bv;
      if (relu_flag) v = v > 0.f ? v : 0.f;
      Ob[(size_t)(rowBase + r) * HD + col] = v;
    }
  }
}

// ----------------------- latent = h2 @ ew3.T + eb3 -------------------------
__global__ __launch_bounds__(256) void latent_kernel(const float* __restrict__ H2,
                                                     const float* __restrict__ ew3,
                                                     const float* __restrict__ eb3,
                                                     float* __restrict__ LAT) {
  size_t i = (size_t)blockIdx.x * 256 + threadIdx.x;      // over B*N*3
  if (i >= (size_t)BATCH * NN * 3) return;
  int j = (int)(i % 3);
  size_t bn = i / 3;
  const float* h = H2 + bn * HD;
  float s = eb3[j];
  for (int k = 0; k < HD; ++k) s += h[k] * ew3[j * HD + k];
  LAT[i] = s;
}

// ----------------- edge lengths + degree + ea segment sum ------------------
__global__ __launch_bounds__(256) void edge_ea_kernel(const int* __restrict__ EI,
                                                      const float* __restrict__ LAT,
                                                      float* __restrict__ EA,
                                                      float* __restrict__ DEG,
                                                      float* __restrict__ EASUM) {
  size_t i = (size_t)blockIdx.x * 256 + threadIdx.x;      // over B*E
  if (i >= (size_t)BATCH * NE) return;
  int b = (int)(i / NE), e = (int)(i % NE);
  int src = EI[(size_t)b * 2 * NE + e];
  int dst = EI[(size_t)b * 2 * NE + NE + e];
  const float* ps = LAT + ((size_t)b * NN + src) * 3;
  const float* pd = LAT + ((size_t)b * NN + dst) * 3;
  float dx = ps[0] - pd[0], dy = ps[1] - pd[1];
  float ea = sqrtf(dx * dx + dy * dy);
  EA[i] = ea;
  atomicAdd(&DEG[(size_t)b * NN + dst], 1.0f);
  atomicAdd(&EASUM[(size_t)b * NN + dst], ea);
}

__global__ __launch_bounds__(256) void loop_ea_kernel(const float* __restrict__ EASUM,
                                                      const float* __restrict__ DEG,
                                                      float* __restrict__ LEA) {
  size_t i = (size_t)blockIdx.x * 256 + threadIdx.x;      // over B*N
  if (i >= (size_t)BATCH * NN) return;
  LEA[i] = EASUM[i] / fmaxf(DEG[i], 1.0f);
}

// --------------------- layer-1 rank-1 linear (F_in = 1) --------------------
__global__ __launch_bounds__(256) void lin1_kernel(const float* __restrict__ LAT,
                                                   const float* __restrict__ wl,
                                                   const float* __restrict__ bl,
                                                   const float* __restrict__ wr,
                                                   const float* __restrict__ br,
                                                   float* __restrict__ XL,
                                                   float* __restrict__ XR) {
  size_t i = (size_t)blockIdx.x * 256 + threadIdx.x;      // over B*N*128
  if (i >= BIG) return;
  int h = (int)(i & 127);
  size_t bn = i >> 7;
  float xin = LAT[bn * 3 + 2];
  XL[i] = xin * wl[h] + bl[h];
  XR[i] = xin * wr[h] + br[h];
}

// ------------------- per-edge attention score + seg-max --------------------
template <bool HAS_EA>
__global__ __launch_bounds__(256) void edge_score_kernel(const int* __restrict__ EI,
                                                         const float* __restrict__ XL,
                                                         const float* __restrict__ XR,
                                                         const float* __restrict__ We,
                                                         const float* __restrict__ att,
                                                         const float* __restrict__ EA,
                                                         const float* __restrict__ LEA,
                                                         float* __restrict__ ESC,
                                                         float* __restrict__ EMAX) {
  const int lane = threadIdx.x & 31;
  const int wid  = threadIdx.x >> 5;
  const int b    = blockIdx.y;
  const long edge = (long)blockIdx.x * 8 + wid;           // [0, ET)
  int src, dst; float ea;
  if (edge < NE) {
    src = EI[(size_t)b * 2 * NE + edge];
    dst = EI[(size_t)b * 2 * NE + NE + edge];
    ea  = EA[(size_t)b * NE + edge];
  } else {
    int n = (int)(edge - NE);
    src = dst = n;
    ea  = LEA[(size_t)b * NN + n];
  }
  const float* xl = XL + ((size_t)b * NN + src) * HD;
  const float* xr = XR + ((size_t)b * NN + dst) * HD;
  float s = 0.f;
#pragma unroll
  for (int j = 0; j < 4; ++j) {
    int h = lane + 32 * j;
    float m = xl[h] + xr[h];
    if (HAS_EA) m += ea * We[h];
    float lr = m > 0.f ? m : 0.2f * m;
    s += att[h] * lr;
  }
#pragma unroll
  for (int off = 16; off; off >>= 1) s += __shfl_xor(s, off, 32);
  if (lane == 0) {
    ESC[(size_t)b * ET + edge] = s;
    atomicMaxF(&EMAX[(size_t)b * NN + dst], s);
  }
}

// ------------- per-edge exp + denominator + numerator scatter --------------
__global__ __launch_bounds__(256) void edge_accum_kernel(const int* __restrict__ EI,
                                                         const float* __restrict__ ESC,
                                                         const float* __restrict__ EMAX,
                                                         const float* __restrict__ XL,
                                                         float* __restrict__ DEN,
                                                         float* __restrict__ NUM) {
  const int lane = threadIdx.x & 31;
  const int wid  = threadIdx.x >> 5;
  const int b    = blockIdx.y;
  const long edge = (long)blockIdx.x * 8 + wid;
  int src, dst;
  if (edge < NE) {
    src = EI[(size_t)b * 2 * NE + edge];
    dst = EI[(size_t)b * 2 * NE + NE + edge];
  } else {
    src = dst = (int)(edge - NE);
  }
  float ex = __expf(ESC[(size_t)b * ET + edge] - EMAX[(size_t)b * NN + dst]);
  if (lane == 0) atomicAdd(&DEN[(size_t)b * NN + dst], ex);
  const float* xl = XL + ((size_t)b * NN + src) * HD;
  float* num = NUM + ((size_t)b * NN + dst) * HD;
#pragma unroll
  for (int j = 0; j < 4; ++j) {
    int h = lane + 32 * j;
    atomicAdd(&num[h], ex * xl[h]);
  }
}

// ----------------------------- node finalize -------------------------------
__global__ __launch_bounds__(256) void finalize_plain_kernel(float* __restrict__ X,
                                                             const float* __restrict__ DEN,
                                                             const float* __restrict__ bias) {
  size_t i = (size_t)blockIdx.x * 256 + threadIdx.x;
  if (i >= BIG) return;
  int h = (int)(i & 127);
  size_t bn = i >> 7;
  float v = X[i] / DEN[bn] + bias[h];
  X[i] = v > 0.f ? v : 0.f;
}

__global__ __launch_bounds__(256) void finalize_skip_kernel(float* __restrict__ X,
                                                            const float* __restrict__ DEN,
                                                            const float* __restrict__ bias,
                                                            const float* __restrict__ LAT,
                                                            const float* __restrict__ sw,
                                                            const float* __restrict__ sb) {
  size_t i = (size_t)blockIdx.x * 256 + threadIdx.x;
  if (i >= BIG) return;
  int h = (int)(i & 127);
  size_t bn = i >> 7;
  const float* l = LAT + bn * 3;
  float skip = sb[h];
#pragma unroll
  for (int k = 0; k < 3; ++k) skip += l[k] * sw[h * 3 + k];
  float v = X[i] / DEN[bn] + bias[h] + 0.1f * skip;
  X[i] = v > 0.f ? v : 0.f;
}

// -------------------------------- heads ------------------------------------
__global__ __launch_bounds__(256) void head4_kernel(const float* __restrict__ X,
                                                    const float* __restrict__ lw,
                                                    const float* __restrict__ lb,
                                                    float* __restrict__ out) {
  size_t i = (size_t)blockIdx.x * 256 + threadIdx.x;      // over B*N*4
  if (i >= (size_t)BATCH * NN * 4) return;
  int j = (int)(i & 3);
  size_t bn = i >> 2;
  const float* x = X + bn * HD;
  float s = lb[j];
  for (int k = 0; k < HD; ++k) s += x[k] * lw[j * HD + k];
  out[i] = s;
}

__global__ __launch_bounds__(256) void head1_kernel(const float* __restrict__ X,
                                                    const float* __restrict__ vw,
                                                    const float* __restrict__ vb,
                                                    float* __restrict__ out) {
  size_t i = (size_t)blockIdx.x * 256 + threadIdx.x;      // over B*N
  if (i >= (size_t)BATCH * NN) return;
  const float* x = X + i * HD;
  float s = vb[0];
  for (int k = 0; k < HD; ++k) s += x[k] * vw[k];
  out[i] = s;
}

// ------------------------------- launcher ----------------------------------
static inline dim3 g1d(size_t n) { return dim3((unsigned)((n + 255) / 256)); }

extern "C" void kernel_launch(void* const* d_in, const int* in_sizes, int n_in,
                              void* d_out, int out_size, void* d_ws, size_t ws_size,
                              hipStream_t stream) {
  (void)in_sizes; (void)n_in; (void)out_size; (void)ws_size;
  const float* obs  = (const float*)d_in[0];
  const int*   ei   = (const int*)  d_in[1];
  const float* ew1 = (const float*)d_in[2],  *eb1 = (const float*)d_in[3];
  const float* ew2 = (const float*)d_in[4],  *eb2 = (const float*)d_in[5];
  const float* ew3 = (const float*)d_in[6],  *eb3 = (const float*)d_in[7];
  const float* g1wl= (const float*)d_in[8],  *g1bl= (const float*)d_in[9];
  const float* g1wr= (const float*)d_in[10], *g1br= (const float*)d_in[11];
  const float* g1we= (const float*)d_in[12], *g1att=(const float*)d_in[13], *g1b=(const float*)d_in[14];
  const float* g2wl= (const float*)d_in[15], *g2bl= (const float*)d_in[16];
  const float* g2wr= (const float*)d_in[17], *g2br= (const float*)d_in[18];
  const float* g2we= (const float*)d_in[19], *g2att=(const float*)d_in[20], *g2b=(const float*)d_in[21];
  const float* g3wl= (const float*)d_in[22], *g3bl= (const float*)d_in[23];
  const float* g3wr= (const float*)d_in[24], *g3br= (const float*)d_in[25];
  const float* g3att=(const float*)d_in[26], *g3b = (const float*)d_in[27];
  const float* g4wl= (const float*)d_in[28], *g4bl= (const float*)d_in[29];
  const float* g4wr= (const float*)d_in[30], *g4br= (const float*)d_in[31];
  const float* g4att=(const float*)d_in[32], *g4b = (const float*)d_in[33];
  const float* lw  = (const float*)d_in[34], *lb  = (const float*)d_in[35];
  const float* vw  = (const float*)d_in[36], *vb  = (const float*)d_in[37];
  const float* sw  = (const float*)d_in[38], *sb  = (const float*)d_in[39];
  float* out = (float*)d_out;

  // workspace layout (fp32 elements)
  float* ws = (float*)d_ws;
  size_t off = 0;
  float* LAT = ws + off; off += (size_t)BATCH * NN * 3;   // latent
  float* DEG = ws + off; off += (size_t)BATCH * NN;       // degree, reused as DEN
  float* EMX = ws + off; off += (size_t)BATCH * NN;       // ea-sum, reused as EMAX
  float* LEA = ws + off; off += (size_t)BATCH * NN;       // self-loop edge attr
  float* EA  = ws + off; off += (size_t)BATCH * NE;       // edge lengths
  float* ESC = ws + off; off += (size_t)BATCH * ET;       // per-edge scores
  float* XL  = ws + off; off += BIG;                      // also h1
  float* XR  = ws + off; off += BIG;                      // also h2
  float* X1  = ws + off; off += BIG;
  float* X2  = ws + off; off += BIG;
  float* X3  = ws + off; off += BIG;                      // also x4

  const float NEG_INF = -__builtin_inff();
  const dim3 BLK(256);
  const dim3 GEMM_G(NN / 128, BATCH);
  const dim3 EDGE_G(ET / 8, BATCH);

  // ---- encoder ----
  h1_kernel<<<g1d(BIG), BLK, 0, stream>>>(obs, ew1, eb1, XL);
  gemm128_kernel<<<GEMM_G, BLK, 0, stream>>>(XL, ew2, eb2, XR, 1);          // h2
  latent_kernel<<<g1d((size_t)BATCH * NN * 3), BLK, 0, stream>>>(XR, ew3, eb3, LAT);

  // ---- edge attrs + self loops ----
  fill_kernel<<<g1d((size_t)BATCH * NN), BLK, 0, stream>>>(DEG, 0.f, (size_t)BATCH * NN);
  fill_kernel<<<g1d((size_t)BATCH * NN), BLK, 0, stream>>>(EMX, 0.f, (size_t)BATCH * NN);
  edge_ea_kernel<<<g1d((size_t)BATCH * NE), BLK, 0, stream>>>(ei, LAT, EA, DEG, EMX);
  loop_ea_kernel<<<g1d((size_t)BATCH * NN), BLK, 0, stream>>>(EMX, DEG, LEA);

  auto run_gat = [&](const float* XLp, const float* XRp, const float* Wep,
                     const float* attp, float* Xout, bool has_ea) {
    fill_kernel<<<g1d((size_t)BATCH * NN), BLK, 0, stream>>>(EMX, NEG_INF, (size_t)BATCH * NN);
    fill_kernel<<<g1d((size_t)BATCH * NN), BLK, 0, stream>>>(DEG, 0.f, (size_t)BATCH * NN);
    fill_kernel<<<g1d(BIG), BLK, 0, stream>>>(Xout, 0.f, BIG);
    if (has_ea)
      edge_score_kernel<true><<<EDGE_G, BLK, 0, stream>>>(ei, XLp, XRp, Wep, attp, EA, LEA, ESC, EMX);
    else
      edge_score_kernel<false><<<EDGE_G, BLK, 0, stream>>>(ei, XLp, XRp, Wep, attp, EA, LEA, ESC, EMX);
    edge_accum_kernel<<<EDGE_G, BLK, 0, stream>>>(ei, ESC, EMX, XLp, DEG, Xout);
  };

  // ---- GAT layer 1 (F_in = 1, has edge attrs) ----
  lin1_kernel<<<g1d(BIG), BLK, 0, stream>>>(LAT, g1wl, g1bl, g1wr, g1br, XL, XR);
  run_gat(XL, XR, g1we, g1att, X1, true);
  finalize_plain_kernel<<<g1d(BIG), BLK, 0, stream>>>(X1, DEG, g1b);

  // ---- GAT layer 2 (has edge attrs) ----
  gemm128_kernel<<<GEMM_G, BLK, 0, stream>>>(X1, g2wl, g2bl, XL, 0);
  gemm128_kernel<<<GEMM_G, BLK, 0, stream>>>(X1, g2wr, g2br, XR, 0);
  run_gat(XL, XR, g2we, g2att, X2, true);
  finalize_plain_kernel<<<g1d(BIG), BLK, 0, stream>>>(X2, DEG, g2b);

  // ---- GAT layer 3 (+ skip) -> logits ----
  gemm128_kernel<<<GEMM_G, BLK, 0, stream>>>(X2, g3wl, g3bl, XL, 0);
  gemm128_kernel<<<GEMM_G, BLK, 0, stream>>>(X2, g3wr, g3br, XR, 0);
  run_gat(XL, XR, nullptr, g3att, X3, false);
  finalize_skip_kernel<<<g1d(BIG), BLK, 0, stream>>>(X3, DEG, g3b, LAT, sw, sb);
  head4_kernel<<<g1d((size_t)BATCH * NN * 4), BLK, 0, stream>>>(X3, lw, lb, out);

  // ---- GAT layer 4 (+ skip) -> vals (reuses X3) ----
  gemm128_kernel<<<GEMM_G, BLK, 0, stream>>>(X2, g4wl, g4bl, XL, 0);
  gemm128_kernel<<<GEMM_G, BLK, 0, stream>>>(X2, g4wr, g4br, XR, 0);
  run_gat(XL, XR, nullptr, g4att, X3, false);
  finalize_skip_kernel<<<g1d(BIG), BLK, 0, stream>>>(X3, DEG, g4b, LAT, sw, sb);
  head1_kernel<<<g1d((size_t)BATCH * NN), BLK, 0, stream>>>(X3, vw, vb, out + (size_t)BATCH * NN * 4);
}
